// SelfAttention_7438883357213
// MI455X (gfx1250) — matrix operand order repaired and checked by reading, not compiled, not run on hardware
//
#include <hip/hip_runtime.h>

typedef __attribute__((ext_vector_type(16))) __bf16 v16bf;
typedef __attribute__((ext_vector_type(8)))  __bf16 v8bf;
typedef __attribute__((ext_vector_type(8)))  float  v8f;

#define B_  16
#define T_  256
#define C_  4096
#define H_  16
#define D_  256
#define N3C 12288

__device__ __forceinline__ v8f vzero8() {
  v8f z;
#pragma unroll
  for (int i = 0; i < 8; ++i) z[i] = 0.0f;
  return z;
}

__device__ __forceinline__ v8f wmma_bf16(v16bf a, v16bf b, v8f c) {
  // D = A(16x32 bf16) * B(32x16 bf16) + C(16x16 f32)
  return __builtin_amdgcn_wmma_f32_16x16x32_bf16(false, a, false, b, (short)0, c,
                                                 false, false);
}

// A fragment (16x32 bf16), source row-major [M][ld].
// lane L: row = m0 + (L&15); half = L>>4
// elems 0..7  -> K = k0 + 8*half + e        (contiguous 16B)
// elems 8..15 -> K = k0 + 16 + 8*half + e   (contiguous 16B)
__device__ __forceinline__ v16bf load_a_bf16(const __bf16* base, int ld, int m0,
                                             int k0, int lane) {
  const __bf16* p = base + (size_t)(m0 + (lane & 15)) * ld + k0 + 8 * (lane >> 4);
  v8bf lo = *(const v8bf*)p;
  v8bf hi = *(const v8bf*)(p + 16);
  v16bf a;
#pragma unroll
  for (int e = 0; e < 8; ++e) { a[e] = lo[e]; a[e + 8] = hi[e]; }
  return a;
}

// Same A fragment but converting fp32 source on load.
__device__ __forceinline__ v16bf load_a_f32(const float* base, int ld, int m0,
                                            int k0, int lane) {
  const float* p = base + (size_t)(m0 + (lane & 15)) * ld + k0 + 8 * (lane >> 4);
  v16bf a;
#pragma unroll
  for (int e = 0; e < 8; ++e) { a[e] = (__bf16)p[e]; a[e + 8] = (__bf16)p[16 + e]; }
  return a;
}

// B fragment (32x16 bf16) with B[k][n] = src[n][k], src row-major [N][ld]
// lane L: n = n0 + (L&15); elems 0..7 -> K = k0 + 8*(L>>4) + e, elems 8..15 -> K = k0 + 16 + 8*(L>>4) + e
__device__ __forceinline__ v16bf load_bt_bf16(const __bf16* base, int ld, int n0,
                                              int k0, int lane) {
  const __bf16* p = base + (size_t)(n0 + (lane & 15)) * ld + k0 + 8 * (lane >> 4);
  v8bf lo = *(const v8bf*)p;
  v8bf hi = *(const v8bf*)(p + 16);
  v16bf b;
#pragma unroll
  for (int e = 0; e < 8; ++e) { b[e] = lo[e]; b[e + 8] = hi[e]; }
  return b;
}

// ---------------------------------------------------------------------------
// Transpose + fp32->bf16: WT[n][k] = W[k][n]   (W is [K][N] row-major)
// ---------------------------------------------------------------------------
__global__ __launch_bounds__(256) void k_cvt_transpose(const float* __restrict__ W,
                                                       __bf16* __restrict__ WT,
                                                       int K, int N) {
  size_t total = (size_t)K * N;
  for (size_t idx = blockIdx.x * (size_t)blockDim.x + threadIdx.x; idx < total;
       idx += (size_t)gridDim.x * blockDim.x) {
    size_t n = idx / (size_t)K;
    size_t k = idx - n * (size_t)K;
    WT[idx] = (__bf16)W[k * (size_t)N + n];
  }
}

// ---------------------------------------------------------------------------
// QKV GEMM: [4096,4096](fp32 x) @ WT_attn(bf16, [12288][4096]) + bias
// Each wave: one 16(M) x 128(N) strip. All operand loads of a k-step are
// hoisted into registers (one load clause, staggered waitcnts); unroll-by-2
// lets the scheduler overlap next-step loads with current WMMAs via renaming.
// ---------------------------------------------------------------------------
__global__ __launch_bounds__(256) void k_qkv(const float* __restrict__ x,
                                             const __bf16* __restrict__ WT,
                                             const float* __restrict__ bias,
                                             __bf16* __restrict__ qh,
                                             __bf16* __restrict__ kh,
                                             __bf16* __restrict__ vh) {
  int lane = threadIdx.x & 31;
  int wave = threadIdx.x >> 5;
  int job  = blockIdx.x * 8 + wave;   // 256 M-tiles * 96 N128-blocks = 24576 jobs
  int mt   = job / 96;
  int nb   = job % 96;
  int m0 = mt * 16, n0 = nb * 128;

  v8f acc[8];
#pragma unroll
  for (int j = 0; j < 8; ++j) acc[j] = vzero8();

#pragma unroll 2
  for (int k = 0; k < C_; k += 32) {
    v16bf a = load_a_f32(x, C_, m0, k, lane);
    v16bf b[8];
#pragma unroll
    for (int j = 0; j < 8; ++j) b[j] = load_bt_bf16(WT, C_, n0 + j * 16, k, lane);
    __builtin_prefetch(x + (size_t)(m0 + (lane & 15)) * C_ + k + 64, 0, 1);
#pragma unroll
    for (int j = 0; j < 8; ++j) acc[j] = wmma_bf16(a, b[j], acc[j]);
  }

  int lm = lane & 15, lh = lane >> 4;
#pragma unroll
  for (int j = 0; j < 8; ++j) {
    int n = n0 + j * 16 + lm;
    float bv = bias[n];
    int which = n >> 12;            // 0=q, 1=k, 2=v
    int c = n & (C_ - 1);
    int h = c >> 8, d = c & 255;
    __bf16* outp = (which == 0) ? qh : (which == 1) ? kh : vh;
#pragma unroll
    for (int r = 0; r < 8; ++r) {
      int m = m0 + r + 8 * lh;      // m = b*T + t
      int b = m >> 8, t = m & 255;
      outp[(((size_t)(b * H_ + h) * T_ + t) << 8) + d] = (__bf16)(acc[j][r] + bv);
    }
  }
}

// ---------------------------------------------------------------------------
// Attention: one workgroup (4 waves) per (b,h).
// LDS: K[256][256] bf16 (128KB) + V^T[256][256] bf16 (128KB) + 4 strip bufs (32KB)
// Per wave strip (16 rows): S = relu(mask(QK^T/16)) -> A1 = S@V -> Y = A1@V
// ---------------------------------------------------------------------------
__global__ __launch_bounds__(128) void k_attn(const __bf16* __restrict__ qh,
                                              const __bf16* __restrict__ kh,
                                              const __bf16* __restrict__ vh,
                                              __bf16* __restrict__ yatt) {
  extern __shared__ __align__(32) char smem[];
  __bf16* ksm   = (__bf16*)smem;          // [256][256]
  __bf16* vtsm  = ksm + 65536;            // [256][256], V transposed (vT[d][t])
  __bf16* sbase = vtsm + 65536;           // 4 x [16][256] per-wave strips

  int bh = blockIdx.x;                    // b*H + h
  const __bf16* Qg = qh + (size_t)bh * 65536;
  const __bf16* Kg = kh + (size_t)bh * 65536;
  const __bf16* Vg = vh + (size_t)bh * 65536;

  int tid = threadIdx.x, wave = tid >> 5, lane = tid & 31;

  // Stage K row-major and V transposed into LDS.
  {
    const uint4* src = (const uint4*)Kg;
    uint4* dst = (uint4*)ksm;
    for (int i = tid; i < 8192; i += 128) dst[i] = src[i];
    for (int i = tid; i < 8192; i += 128) {
      int t  = i >> 5;
      int d0 = (i & 31) * 8;
      v8bf v = *(const v8bf*)(Vg + t * 256 + d0);
#pragma unroll
      for (int e = 0; e < 8; ++e) vtsm[(size_t)(d0 + e) * 256 + t] = v[e];
    }
  }
  __syncthreads();

  __bf16* strip = sbase + wave * (16 * 256);
  int lm = lane & 15, lh = lane >> 4;
  int b_ = bh >> 4, h_ = bh & 15;

  for (int mt = wave; mt < 16; mt += 4) {
    int m0 = mt * 16;
    v16bf afrag[8];
    v16bf bfr[8];

    // ---- S = relu(causal_mask(Q K^T * D^-0.5)) -> bf16 strip in LDS ----
#pragma unroll
    for (int kk = 0; kk < 8; ++kk) afrag[kk] = load_a_bf16(Qg, 256, m0, kk * 32, lane);
    for (int nt = 0; nt < 16; ++nt) {
#pragma unroll
      for (int kk = 0; kk < 8; ++kk) bfr[kk] = load_bt_bf16(ksm, 256, nt * 16, kk * 32, lane);
      v8f acc = vzero8();
#pragma unroll
      for (int kk = 0; kk < 8; ++kk) acc = wmma_bf16(afrag[kk], bfr[kk], acc);
      int n = nt * 16 + lm;
#pragma unroll
      for (int r = 0; r < 8; ++r) {
        int m = m0 + r + 8 * lh;
        float s = acc[r] * (1.0f / 16.0f);
        s = (n <= m) ? fmaxf(s, 0.0f) : 0.0f;  // mask -> -inf, then relu => 0
        strip[(r + 8 * lh) * 256 + n] = (__bf16)s;
      }
    }
    asm volatile("s_wait_dscnt 0" ::: "memory");

    // ---- A1 = S @ V : S fully cached in afrag registers, overwrite strip ----
#pragma unroll
    for (int kk = 0; kk < 8; ++kk) afrag[kk] = load_a_bf16(strip, 256, 0, kk * 32, lane);
    for (int dt = 0; dt < 16; ++dt) {
#pragma unroll
      for (int kk = 0; kk < 8; ++kk) bfr[kk] = load_bt_bf16(vtsm, 256, dt * 16, kk * 32, lane);
      v8f acc = vzero8();
#pragma unroll
      for (int kk = 0; kk < 8; ++kk) acc = wmma_bf16(afrag[kk], bfr[kk], acc);
      int d = dt * 16 + lm;
#pragma unroll
      for (int r = 0; r < 8; ++r)
        strip[(r + 8 * lh) * 256 + d] = (__bf16)acc[r];
    }
    asm volatile("s_wait_dscnt 0" ::: "memory");

    // ---- Y = A1 @ V -> yatt[b][t][h*256+d] ----
#pragma unroll
    for (int kk = 0; kk < 8; ++kk) afrag[kk] = load_a_bf16(strip, 256, 0, kk * 32, lane);
    for (int dt = 0; dt < 16; ++dt) {
#pragma unroll
      for (int kk = 0; kk < 8; ++kk) bfr[kk] = load_bt_bf16(vtsm, 256, dt * 16, kk * 32, lane);
      v8f acc = vzero8();
#pragma unroll
      for (int kk = 0; kk < 8; ++kk) acc = wmma_bf16(afrag[kk], bfr[kk], acc);
      int d = dt * 16 + lm;
#pragma unroll
      for (int r = 0; r < 8; ++r) {
        int t = m0 + r + 8 * lh;
        yatt[((size_t)(b_ * T_ + t)) * C_ + h_ * 256 + d] = (__bf16)acc[r];
      }
    }
  }
}

// ---------------------------------------------------------------------------
// Projection GEMM: yatt(bf16 [4096][4096]) @ WT_proj(bf16 [4096][4096]) + bias
// ---------------------------------------------------------------------------
__global__ __launch_bounds__(256) void k_proj(const __bf16* __restrict__ yatt,
                                              const __bf16* __restrict__ WT,
                                              const float* __restrict__ bias,
                                              float* __restrict__ out) {
  int lane = threadIdx.x & 31;
  int wave = threadIdx.x >> 5;
  int job  = blockIdx.x * 8 + wave;   // 256 * 32 = 8192 jobs
  int mt   = job / 32;
  int nb   = job % 32;
  int m0 = mt * 16, n0 = nb * 128;

  v8f acc[8];
#pragma unroll
  for (int j = 0; j < 8; ++j) acc[j] = vzero8();

#pragma unroll 2
  for (int k = 0; k < C_; k += 32) {
    v16bf a = load_a_bf16(yatt, C_, m0, k, lane);
    v16bf b[8];
#pragma unroll
    for (int j = 0; j < 8; ++j) b[j] = load_bt_bf16(WT, C_, n0 + j * 16, k, lane);
#pragma unroll
    for (int j = 0; j < 8; ++j) acc[j] = wmma_bf16(a, b[j], acc[j]);
  }

  int lm = lane & 15, lh = lane >> 4;
#pragma unroll
  for (int j = 0; j < 8; ++j) {
    int n = n0 + j * 16 + lm;
    float bv = bias[n];
#pragma unroll
    for (int r = 0; r < 8; ++r) {
      int m = m0 + r + 8 * lh;
      out[(size_t)m * C_ + n] = acc[j][r] + bv;
    }
  }
}

// ---------------------------------------------------------------------------
extern "C" void kernel_launch(void* const* d_in, const int* in_sizes, int n_in,
                              void* d_out, int out_size, void* d_ws, size_t ws_size,
                              hipStream_t stream) {
  (void)in_sizes; (void)n_in; (void)out_size; (void)ws_size;
  const float* x      = (const float*)d_in[0];
  const float* W_attn = (const float*)d_in[1];
  const float* b_attn = (const float*)d_in[2];
  const float* W_proj = (const float*)d_in[3];
  const float* b_proj = (const float*)d_in[4];
  float* out = (float*)d_out;

  char* ws = (char*)d_ws;
  __bf16* WTa  = (__bf16*)(ws);                    // 12288*4096 bf16 = 96 MiB
  __bf16* WTp  = (__bf16*)(ws + 100663296ull);     // 4096*4096  bf16 = 32 MiB
  __bf16* qh   = (__bf16*)(ws + 134217728ull);     // [B,H,T,D]  bf16 = 32 MiB
  __bf16* kh   = (__bf16*)(ws + 167772160ull);     // 32 MiB
  __bf16* vh   = (__bf16*)(ws + 201326592ull);     // 32 MiB
  __bf16* yatt = (__bf16*)(ws + 234881024ull);     // [B,T,C]    bf16 = 32 MiB

  k_cvt_transpose<<<2048, 256, 0, stream>>>(W_attn, WTa, C_, N3C);
  k_cvt_transpose<<<1024, 256, 0, stream>>>(W_proj, WTp, C_, C_);
  k_qkv<<<3072, 256, 0, stream>>>(x, WTa, b_attn, qh, kh, vh);
  k_attn<<<256, 128, 294912, stream>>>(qh, kh, vh, yatt);  // 288 KB dynamic LDS
  k_proj<<<1024, 256, 0, stream>>>(yatt, WTp, b_proj, out);
}
